// GAT_61701500174377
// MI455X (gfx1250) — matrix operand order repaired and checked by reading, not compiled, not run on hardware
//
#include <hip/hip_runtime.h>
#include <hip/hip_bf16.h>
#include <math.h>

// ---------------- problem constants (from reference) ----------------
#define F_IN      165
#define KP0       192      // F_IN padded to multiple of 32 for 16x16x32 WMMA
#define HEADS     4
#define HID       64
#define NC        256      // HEADS*HID
#define NCP2      16       // layer-2 output cols (2) padded to one 16-col tile
#define NEG_SLOPE 0.2f
#define BN_EPS    1e-5f

typedef __attribute__((ext_vector_type(16))) __bf16 v16bf;
typedef __attribute__((ext_vector_type(8)))  float  v8f;

union Frag16 { v16bf v; uint4 q[2]; };
union FragC  { v8f  v; float f[8]; };

// f32 -> bf16 round-to-nearest-even
__device__ __forceinline__ unsigned short f32_to_bf16(float f) {
    unsigned int x = __float_as_uint(f);
    unsigned int r = x + 0x7fffu + ((x >> 16) & 1u);
    return (unsigned short)(r >> 16);
}

// float atomic max via monotonic int mapping (memory init'd to -inf)
__device__ __forceinline__ void atomicMaxF(float* addr, float val) {
    if (val >= 0.0f) atomicMax((int*)addr, __float_as_int(val));
    else             atomicMin((unsigned int*)addr, __float_as_uint(val));
}

// ---------------- generic utility kernels ----------------
__global__ void zero_f32(float* p, long long n) {
    long long i = (long long)blockIdx.x * blockDim.x + threadIdx.x;
    if (i < n) p[i] = 0.0f;
}

// x f32 [Nn,K] -> A bf16 [Nn,Kp] (zero-padded K)
__global__ void convert_pad(const float* __restrict__ x, unsigned short* __restrict__ A,
                            int Nn, int K, int Kp) {
    long long i = (long long)blockIdx.x * blockDim.x + threadIdx.x;
    long long total = (long long)Nn * Kp;
    if (i >= total) return;
    int k = (int)(i % Kp);
    long long n = i / Kp;
    float v = (k < K) ? x[n * K + k] : 0.0f;
    A[i] = f32_to_bf16(v);
}

// W f32 [K,Ncol] -> WT bf16 [NcolP,Kp]  (transpose + pad both dims)
__global__ void convert_w(const float* __restrict__ W, unsigned short* __restrict__ WT,
                          int K, int Kp, int Ncol, int NcolP) {
    long long i = (long long)blockIdx.x * blockDim.x + threadIdx.x;
    long long total = (long long)NcolP * Kp;
    if (i >= total) return;
    int k = (int)(i % Kp);
    int c = (int)(i / Kp);
    float v = (k < K && c < Ncol) ? W[(long long)k * Ncol + c] : 0.0f;
    WT[i] = f32_to_bf16(v);
}

// ---------------- WMMA GEMM: C[M,Ncol] = A[M,Kp](bf16) @ WT[Ncol,Kp]^T(bf16) ----------------
// one wave computes a 16 x (16*NT) strip: A fragment loaded once per k-step,
// reused across NT column tiles (NT accumulators in registers).
// M multiple of 16, Kp multiple of 32, Ncol multiple of 16*NT.
template <int NT>
__global__ __launch_bounds__(32) void gemm_bf16_wmma(
    const unsigned short* __restrict__ A,
    const unsigned short* __restrict__ BT,
    float* __restrict__ C, int Kp, int Ncol)
{
    const int lane = threadIdx.x;
    const int l15  = lane & 15;
    const int half = lane >> 4;
    const int row  = blockIdx.y * 16 + l15;
    const int col0 = blockIdx.x * (16 * NT) + l15;

    const unsigned short* arow = A + (long long)row * Kp;
    const unsigned short* brow[NT];
#pragma unroll
    for (int t = 0; t < NT; ++t) brow[t] = BT + (long long)(col0 + t * 16) * Kp;

    FragC acc[NT];
#pragma unroll
    for (int t = 0; t < NT; ++t)
#pragma unroll
        for (int i = 0; i < 8; ++i) acc[t].f[i] = 0.0f;

    for (int k0 = 0; k0 < Kp; k0 += 32) {
        Frag16 a;
        // A fragment: lanes 0-15 K={k0..k0+7, k0+16..k0+23}; lanes 16-31 shifted by 8
        a.q[0] = *(const uint4*)(arow + k0 + half * 8);
        a.q[1] = *(const uint4*)(arow + k0 + 16 + half * 8);
        __builtin_prefetch((const void*)(arow + k0 + 32), 0, 3);  // WGP-scope streaming hint
#pragma unroll
        for (int t = 0; t < NT; ++t) {
            Frag16 b;
            // B fragment (from transposed weights): same K order as A fragment
            b.q[0] = *(const uint4*)(brow[t] + k0 + half * 8);
            b.q[1] = *(const uint4*)(brow[t] + k0 + 16 + half * 8);
            acc[t].v = __builtin_amdgcn_wmma_f32_16x16x32_bf16(
                false, a.v, false, b.v, (short)0, acc[t].v, false, false);
        }
    }
    // D layout: VGPR v holds (M = tileM*16 + half*8 + v, N = col)
    float* crow = C + (long long)(blockIdx.y * 16 + half * 8) * Ncol + col0;
#pragma unroll
    for (int t = 0; t < NT; ++t)
#pragma unroll
        for (int v = 0; v < 8; ++v) crow[(long long)v * Ncol + t * 16] = acc[t].f[v];
}

// ---------------- attention alpha pre-computation ----------------
// a_src/a_dst[n*H+h] = sum_c h[n,h*C+c] * att_{src,dst}[h*C+c]; init m=-inf, s=0
__global__ void node_att(const float* __restrict__ h, const float* __restrict__ att_src,
                         const float* __restrict__ att_dst, float* __restrict__ a_src,
                         float* __restrict__ a_dst, float* __restrict__ m, float* __restrict__ s,
                         int Nn, int H, int C, int ldh) {
    int idx = blockIdx.x * blockDim.x + threadIdx.x;
    if (idx >= Nn * H) return;
    int n = idx / H, hd = idx % H;
    const float* hp = h + (long long)n * ldh + hd * C;
    float ss = 0.0f, sd = 0.0f;
    for (int c = 0; c < C; ++c) {
        float v = hp[c];
        ss += v * att_src[hd * C + c];
        sd += v * att_dst[hd * C + c];
    }
    a_src[idx] = ss;
    a_dst[idx] = sd;
    m[idx] = -__builtin_huge_valf();
    s[idx] = 0.0f;
}

// e = leakyrelu(a_src[src]+a_dst[dst]); segment max into m
__global__ void edge_pass1(const int* __restrict__ ei, int E, int Nn, int H,
                           const float* __restrict__ a_src, const float* __restrict__ a_dst,
                           float* __restrict__ eatt, float* __restrict__ m) {
    long long idx = (long long)blockIdx.x * blockDim.x + threadIdx.x;
    long long tot = (long long)(E + Nn) * H;
    if (idx >= tot) return;
    int e = (int)(idx / H), hd = (int)(idx % H);
    int src, dst;
    if (e < E) { src = ei[e]; dst = ei[E + e]; } else { src = dst = e - E; }
    float v = a_src[src * H + hd] + a_dst[dst * H + hd];
    v = v > 0.0f ? v : NEG_SLOPE * v;
    eatt[idx] = v;
    atomicMaxF(m + dst * H + hd, v);
}

// ex = exp(e - m[dst]); segment sum into s
__global__ void edge_pass2(const int* __restrict__ ei, int E, int Nn, int H,
                           const float* __restrict__ m, float* __restrict__ eatt,
                           float* __restrict__ s) {
    long long idx = (long long)blockIdx.x * blockDim.x + threadIdx.x;
    long long tot = (long long)(E + Nn) * H;
    if (idx >= tot) return;
    int e = (int)(idx / H), hd = (int)(idx % H);
    int dst = (e < E) ? ei[E + e] : e - E;
    float ex = expf(eatt[idx] - m[dst * H + hd]);
    eatt[idx] = ex;
    atomicAdd(s + dst * H + hd, ex);
}

// weighted scatter-add (H=4, C=64): one 32-lane group per edge, 8 channels/lane
__global__ __launch_bounds__(256) void edge_scatter(
    const int* __restrict__ ei, int E, int Nn, const float* __restrict__ eatt,
    const float* __restrict__ s, const float* __restrict__ h, float* __restrict__ out) {
    int gid  = blockIdx.x * (blockDim.x >> 5) + (threadIdx.x >> 5);
    int lane = threadIdx.x & 31;
    if (gid >= E + Nn) return;
    int src, dst;
    if (gid < E) { src = ei[gid]; dst = ei[E + gid]; } else { src = dst = gid - E; }
    int hd = lane >> 3;                      // 8 lanes per head
    float alpha = eatt[(long long)gid * 4 + hd] / (s[dst * 4 + hd] + 1e-16f);
    const float* hs = h   + (long long)src * NC + lane * 8;
    float*       od = out + (long long)dst * NC + lane * 8;
#pragma unroll
    for (int i = 0; i < 8; ++i) atomicAdd(od + i, alpha * hs[i]);
}

// layer-2 scatter (H=1, C=2, h row stride 16)
__global__ void edge_scatter2(const int* __restrict__ ei, int E, int Nn,
                              const float* __restrict__ eatt, const float* __restrict__ s,
                              const float* __restrict__ h, float* __restrict__ out) {
    int gid = blockIdx.x * blockDim.x + threadIdx.x;
    if (gid >= E + Nn) return;
    int src, dst;
    if (gid < E) { src = ei[gid]; dst = ei[E + gid]; } else { src = dst = gid - E; }
    float alpha = eatt[gid] / (s[dst] + 1e-16f);
    atomicAdd(out + (long long)dst * 2 + 0, alpha * h[(long long)src * 16 + 0]);
    atomicAdd(out + (long long)dst * 2 + 1, alpha * h[(long long)src * 16 + 1]);
}

// out[n,c] += bias[c]; accumulate channel sums / sumsq (256 channels)
__global__ __launch_bounds__(256) void bias_stats(float* __restrict__ out,
                                                  const float* __restrict__ bias,
                                                  float* __restrict__ sum, float* __restrict__ sumsq,
                                                  int Nn, int rowsPerBlock) {
    int c  = threadIdx.x;
    int r0 = blockIdx.x * rowsPerBlock;
    int r1 = min(r0 + rowsPerBlock, Nn);
    float b = bias[c], ps = 0.0f, pq = 0.0f;
    for (int r = r0; r < r1; ++r) {
        float v = out[(long long)r * NC + c] + b;
        out[(long long)r * NC + c] = v;
        ps += v; pq += v * v;
    }
    atomicAdd(sum + c, ps);
    atomicAdd(sumsq + c, pq);
}

// batchnorm + relu + bf16 convert into next layer's A matrix
__global__ void bn_relu_bf16(const float* __restrict__ out, const float* __restrict__ sum,
                             const float* __restrict__ sumsq, const float* __restrict__ gamma,
                             const float* __restrict__ beta, unsigned short* __restrict__ A, int Nn) {
    long long idx = (long long)blockIdx.x * blockDim.x + threadIdx.x;
    if (idx >= (long long)Nn * NC) return;
    int c = (int)(idx & (NC - 1));
    float mu  = sum[c] / (float)Nn;
    float var = sumsq[c] / (float)Nn - mu * mu;
    float v = gamma[c] * (out[idx] - mu) * rsqrtf(var + BN_EPS) + beta[c];
    A[idx] = f32_to_bf16(v > 0.0f ? v : 0.0f);
}

// final: add bias, log_softmax over 2 classes
__global__ void logsoftmax2(const float* __restrict__ out2, const float* __restrict__ bias2,
                            float* __restrict__ y, int Nn) {
    int n = blockIdx.x * blockDim.x + threadIdx.x;
    if (n >= Nn) return;
    float v0 = out2[(long long)n * 2 + 0] + bias2[0];
    float v1 = out2[(long long)n * 2 + 1] + bias2[1];
    float mx = fmaxf(v0, v1);
    float lse = mx + logf(expf(v0 - mx) + expf(v1 - mx));
    y[(long long)n * 2 + 0] = v0 - lse;
    y[(long long)n * 2 + 1] = v1 - lse;
}

// ---------------- host orchestration ----------------
static inline dim3 grid1d(long long n, int b) { return dim3((unsigned)((n + b - 1) / b)); }

extern "C" void kernel_launch(void* const* d_in, const int* in_sizes, int n_in,
                              void* d_out, int out_size, void* d_ws, size_t ws_size,
                              hipStream_t stream) {
    // input order: x, edge_index, l0{W,att_src,att_dst,bias}, bn0{gamma,beta},
    //              l1{W,att_src,att_dst,bias}, bn1{gamma,beta}, l2{W,att_src,att_dst,bias}
    const float* x   = (const float*)d_in[0];
    const int*   ei  = (const int*)d_in[1];
    const float* W0  = (const float*)d_in[2];
    const float* as0 = (const float*)d_in[3];
    const float* ad0 = (const float*)d_in[4];
    const float* b0  = (const float*)d_in[5];
    const float* g0  = (const float*)d_in[6];
    const float* be0 = (const float*)d_in[7];
    const float* W1  = (const float*)d_in[8];
    const float* as1 = (const float*)d_in[9];
    const float* ad1 = (const float*)d_in[10];
    const float* b1  = (const float*)d_in[11];
    const float* g1  = (const float*)d_in[12];
    const float* be1 = (const float*)d_in[13];
    const float* W2  = (const float*)d_in[14];
    const float* as2 = (const float*)d_in[15];
    const float* ad2 = (const float*)d_in[16];
    const float* b2  = (const float*)d_in[17];

    const int N = in_sizes[0] / F_IN;        // 50000 (multiple of 16)
    const int E = in_sizes[1] / 2;           // 800000
    const int Etot = E + N;

    // workspace carve-up (256B aligned)
    size_t off = 0;
    auto carve = [&](size_t bytes) {
        void* p = (char*)d_ws + off;
        off += (bytes + 255) & ~(size_t)255;
        return p;
    };
    unsigned short* Abf = (unsigned short*)carve((size_t)N * NC * 2);   // bf16 activations
    unsigned short* WT  = (unsigned short*)carve((size_t)NC * NC * 2);  // bf16 transposed W
    float* h    = (float*)carve((size_t)N * NC * 4);                    // GEMM output
    float* outp = (float*)carve((size_t)N * NC * 4);                    // aggregation
    float* a_s  = (float*)carve((size_t)N * HEADS * 4);
    float* a_d  = (float*)carve((size_t)N * HEADS * 4);
    float* mseg = (float*)carve((size_t)N * HEADS * 4);
    float* sseg = (float*)carve((size_t)N * HEADS * 4);
    float* eatt = (float*)carve((size_t)Etot * HEADS * 4);
    float* sum  = (float*)carve(256 * 4);
    float* sq   = (float*)carve(256 * 4);

    const int B = 256;

    // ---------------- layer 0: GAT(165 -> 4x64) ----------------
    convert_pad<<<grid1d((long long)N * KP0, B), B, 0, stream>>>(x, Abf, N, F_IN, KP0);
    convert_w<<<grid1d((long long)NC * KP0, B), B, 0, stream>>>(W0, WT, F_IN, KP0, NC, NC);
    gemm_bf16_wmma<4><<<dim3(NC / 64, N / 16), 32, 0, stream>>>(Abf, WT, h, KP0, NC);
    node_att<<<grid1d((long long)N * HEADS, B), B, 0, stream>>>(h, as0, ad0, a_s, a_d, mseg, sseg, N, HEADS, HID, NC);
    zero_f32<<<grid1d((long long)N * NC, B), B, 0, stream>>>(outp, (long long)N * NC);
    edge_pass1<<<grid1d((long long)Etot * HEADS, B), B, 0, stream>>>(ei, E, N, HEADS, a_s, a_d, eatt, mseg);
    edge_pass2<<<grid1d((long long)Etot * HEADS, B), B, 0, stream>>>(ei, E, N, HEADS, mseg, eatt, sseg);
    edge_scatter<<<grid1d((long long)Etot * 32, B), B, 0, stream>>>(ei, E, N, eatt, sseg, h, outp);
    zero_f32<<<grid1d(512, B), B, 0, stream>>>(sum, 512);  // zeros sum+sq (contiguous carve)
    bias_stats<<<grid1d(((long long)N + 31) / 32, 1).x, 256, 0, stream>>>(outp, b0, sum, sq, N, 32);
    bn_relu_bf16<<<grid1d((long long)N * NC, B), B, 0, stream>>>(outp, sum, sq, g0, be0, Abf, N);

    // ---------------- layer 1: GAT(256 -> 4x64) ----------------
    convert_w<<<grid1d((long long)NC * NC, B), B, 0, stream>>>(W1, WT, NC, NC, NC, NC);
    gemm_bf16_wmma<4><<<dim3(NC / 64, N / 16), 32, 0, stream>>>(Abf, WT, h, NC, NC);
    node_att<<<grid1d((long long)N * HEADS, B), B, 0, stream>>>(h, as1, ad1, a_s, a_d, mseg, sseg, N, HEADS, HID, NC);
    zero_f32<<<grid1d((long long)N * NC, B), B, 0, stream>>>(outp, (long long)N * NC);
    edge_pass1<<<grid1d((long long)Etot * HEADS, B), B, 0, stream>>>(ei, E, N, HEADS, a_s, a_d, eatt, mseg);
    edge_pass2<<<grid1d((long long)Etot * HEADS, B), B, 0, stream>>>(ei, E, N, HEADS, mseg, eatt, sseg);
    edge_scatter<<<grid1d((long long)Etot * 32, B), B, 0, stream>>>(ei, E, N, eatt, sseg, h, outp);
    zero_f32<<<grid1d(512, B), B, 0, stream>>>(sum, 512);
    bias_stats<<<grid1d(((long long)N + 31) / 32, 1).x, 256, 0, stream>>>(outp, b1, sum, sq, N, 32);
    bn_relu_bf16<<<grid1d((long long)N * NC, B), B, 0, stream>>>(outp, sum, sq, g1, be1, Abf, N);

    // ---------------- layer 2: GAT(256 -> 1x2) + log_softmax ----------------
    convert_w<<<grid1d((long long)NCP2 * NC, B), B, 0, stream>>>(W2, WT, NC, NC, 2, NCP2);
    gemm_bf16_wmma<1><<<dim3(1, N / 16), 32, 0, stream>>>(Abf, WT, h, NC, NCP2); // h: [N,16], cols 0..1 valid
    node_att<<<grid1d((long long)N, B), B, 0, stream>>>(h, as2, ad2, a_s, a_d, mseg, sseg, N, 1, 2, NCP2);
    zero_f32<<<grid1d((long long)N * 2, B), B, 0, stream>>>(outp, (long long)N * 2);
    edge_pass1<<<grid1d((long long)Etot, B), B, 0, stream>>>(ei, E, N, 1, a_s, a_d, eatt, mseg);
    edge_pass2<<<grid1d((long long)Etot, B), B, 0, stream>>>(ei, E, N, 1, mseg, eatt, sseg);
    edge_scatter2<<<grid1d((long long)Etot, B), B, 0, stream>>>(ei, E, N, eatt, sseg, h, outp);
    logsoftmax2<<<grid1d((long long)N, B), B, 0, stream>>>(outp, b2, (float*)d_out, N);
}
